// SparseMoELanguageModel_53841710022748
// MI455X (gfx1250) — matrix operand order repaired and checked by reading, not compiled, not run on hardware
//
#include <hip/hip_runtime.h>
#include <hip/hip_bf16.h>
#include <math.h>

// ---------------- problem constants ----------------
#define Bq   4
#define Tq   1024
#define Cq   1024
#define Hh   16
#define HDq  64
#define Eq   8
#define NTOK 4096          // B*T
#define CAPq 1024
#define DFFq 4096

typedef __bf16 bf16_t;
typedef __attribute__((ext_vector_type(16))) __bf16 v16bf;
typedef __attribute__((ext_vector_type(8)))  __bf16 v8bf;
typedef __attribute__((ext_vector_type(8)))  float  v8f;

// ---------------- fragment helpers -----------------
// A fragment (16x32, M x K), row-major source, lda = row stride (elements).
// Lane L: m = L&15, hi = L>>4.  elem i<8 -> k = 8*hi+i ; elem i>=8 -> k = 16+8*hi+(i-8)
__device__ inline v16bf frag_a(const bf16_t* __restrict__ base, int lda, int lane) {
    int m = lane & 15, hi = lane >> 4;
    const bf16_t* r = base + (long)m * lda + 8 * hi;
    v8bf x0 = *(const v8bf*)(r);
    v8bf x1 = *(const v8bf*)(r + 16);
    v16bf a;
#pragma unroll
    for (int i = 0; i < 8; i++) { a[i] = x0[i]; a[8 + i] = x1[i]; }
    return a;
}

// B fragment (32x16, K x N) where memory holds W[n, k] row-major (ldw = row stride).
// Lane L: n = L&15, kh = L>>4. elem i<8 -> k = 8*kh+i ; elem i>=8 -> k = 16+8*kh+(i-8)
__device__ inline v16bf frag_b(const bf16_t* __restrict__ W, int ldw, int lane) {
    int n = lane & 15, kh = lane >> 4;
    const bf16_t* r = W + (long)n * ldw + 8 * kh;
    v8bf x0 = *(const v8bf*)(r);
    v8bf x1 = *(const v8bf*)(r + 16);
    v16bf b;
#pragma unroll
    for (int i = 0; i < 8; i++) { b[i] = x0[i]; b[8 + i] = x1[i]; }
    return b;
}

// ---------------- generic bf16 WMMA GEMM -----------
// Out[m,n] = sum_k A[m,k] * W[n,k]  (+bias[n]) (optional exact GELU, optional bf16 out)
// EPI: 0 = f32 out; 1 = f32 out + bias; 2 = bf16 out + bias + gelu
// Wave tile: 32(M) x 64(N); block tile: 128 x 128 (8 waves).
template <int EPI>
__global__ __launch_bounds__(256) void gemm_bf16(
    const bf16_t* __restrict__ A, const bf16_t* __restrict__ W,
    const float* __restrict__ bias, void* __restrict__ Out,
    int M, int Nd, int K,
    long aBatch, long wBatch, long bBatch, long oBatch)
{
    int e = blockIdx.z;
    A += (long)e * aBatch;
    W += (long)e * wBatch;
    if (bias) bias += (long)e * bBatch;

    int lane = threadIdx.x & 31;
    int wave = threadIdx.x >> 5;
    int wm = wave & 3;        // 4 M waves, 32 rows each
    int wn = wave >> 2;       // 2 N waves, 64 cols each
    int m0 = blockIdx.y * 128 + wm * 32;
    int n0 = blockIdx.x * 128 + wn * 64;

    v8f acc[2][4] = {};

    const bf16_t* aBase0 = A + (long)m0 * K;
    const bf16_t* aBase1 = A + (long)(m0 + 16) * K;
    for (int k = 0; k < K; k += 32) {
        v16bf a0 = frag_a(aBase0 + k, K, lane);
        v16bf a1 = frag_a(aBase1 + k, K, lane);
        if (k + 32 < K) {
            // gfx1250 global_prefetch_b8 on next K tile (L2-resident weights)
            __builtin_prefetch(aBase0 + k + 32, 0, 2);
            __builtin_prefetch(W + (long)n0 * K + k + 32, 0, 2);
        }
#pragma unroll
        for (int s = 0; s < 4; s++) {
            v16bf bfr = frag_b(W + (long)(n0 + s * 16) * K + k, K, lane);
            acc[0][s] = __builtin_amdgcn_wmma_f32_16x16x32_bf16(
                false, a0, false, bfr, (short)0, acc[0][s], false, false);
            acc[1][s] = __builtin_amdgcn_wmma_f32_16x16x32_bf16(
                false, a1, false, bfr, (short)0, acc[1][s], false, false);
        }
    }

    int nL = lane & 15, hi = lane >> 4;
#pragma unroll
    for (int mi = 0; mi < 2; mi++) {
#pragma unroll
        for (int s = 0; s < 4; s++) {
            int n = n0 + s * 16 + nL;
            float bv = (EPI >= 1) ? bias[n] : 0.0f;
#pragma unroll
            for (int j = 0; j < 8; j++) {
                int m = m0 + mi * 16 + 8 * hi + j;
                float v = acc[mi][s][j] + bv;
                if (EPI == 2) {
                    v = 0.5f * v * (1.0f + erff(v * 0.70710678118f));
                    ((bf16_t*)Out)[(long)e * oBatch + (long)m * Nd + n] = (bf16_t)v;
                } else {
                    ((float*)Out)[(long)e * oBatch + (long)m * Nd + n] = v;
                }
            }
        }
    }
}

// ---------------- weight conversion ----------------
__global__ __launch_bounds__(256) void cvt_bf16(const float* __restrict__ in,
                                                bf16_t* __restrict__ out, long n) {
    long i = (long)blockIdx.x * 256 + threadIdx.x;
    if (i < n) out[i] = (bf16_t)in[i];
}

// in: [e][Kd][Nd] row-major -> out: [e][Nd][Kd]
__global__ __launch_bounds__(256) void cvt_bf16_tr(const float* __restrict__ in,
                                                   bf16_t* __restrict__ out,
                                                   int Kd, int Nd) {
    long base = (long)blockIdx.z * Kd * Nd;
    int n = blockIdx.x * 32 + (threadIdx.x & 31);
    int k0 = blockIdx.y * 32;
    for (int kk = (threadIdx.x >> 5); kk < 32; kk += 8) {
        int k = k0 + kk;
        out[base + (long)n * Kd + k] = (bf16_t)in[base + (long)k * Nd + n];
    }
}

// ---------------- layernorm (f32 in, bf16 out) -----
__global__ __launch_bounds__(256) void layernorm_bf16(
    const float* __restrict__ x, const float* __restrict__ w,
    const float* __restrict__ b, bf16_t* __restrict__ out)
{
    __shared__ float rs[8], rs2[8];
    long row = blockIdx.x;
    const float* xr = x + row * Cq;
    float s = 0.f, s2 = 0.f;
    for (int c = threadIdx.x; c < Cq; c += 256) { float v = xr[c]; s += v; s2 += v * v; }
    for (int o = 16; o >= 1; o >>= 1) { s += __shfl_xor(s, o, 32); s2 += __shfl_xor(s2, o, 32); }
    int wave = threadIdx.x >> 5;
    if ((threadIdx.x & 31) == 0) { rs[wave] = s; rs2[wave] = s2; }
    __syncthreads();
    if (threadIdx.x == 0) {
        float a = 0.f, a2 = 0.f;
        for (int i = 0; i < 8; i++) { a += rs[i]; a2 += rs2[i]; }
        rs[0] = a; rs2[0] = a2;
    }
    __syncthreads();
    float mu = rs[0] * (1.0f / Cq);
    float var = rs2[0] * (1.0f / Cq) - mu * mu;
    float inv = rsqrtf(var + 1e-5f);
    for (int c = threadIdx.x; c < Cq; c += 256) {
        float v = (xr[c] - mu) * inv;
        v = fminf(fmaxf(v, -65000.f), 65000.f);
        out[row * Cq + c] = (bf16_t)(v * w[c] + b[c]);
    }
}

// ---------------- RoPE + split + V transpose -------
// qkv f32 [B,T,3C] viewed as [B,T,H,3*HD]; q/k -> bf16 [B,H,T,HD]; v -> bf16 [B,H,HD,T]
__global__ __launch_bounds__(32) void rope_split(
    const float* __restrict__ qkv, bf16_t* __restrict__ qo,
    bf16_t* __restrict__ ko, bf16_t* __restrict__ vto)
{
    int t = blockIdx.x, h = blockIdx.y, b = blockIdx.z;
    int d = threadIdx.x; // 0..31
    const float* base = qkv + ((long)(b * Tq + t) * Hh + h) * (3 * HDq);
    float inv_freq = __powf(10000.f, -(float)d * (1.0f / 32.0f));
    float sn, cs;
    __sincosf((float)t * inv_freq, &sn, &cs);
    float q1 = base[d], q2 = base[d + 32];
    float k1 = base[HDq + d], k2 = base[HDq + d + 32];
    long qk = ((long)(b * Hh + h) * Tq + t) * HDq;
    qo[qk + d]      = (bf16_t)(q1 * cs - q2 * sn);
    qo[qk + d + 32] = (bf16_t)(q1 * sn + q2 * cs);
    ko[qk + d]      = (bf16_t)(k1 * cs - k2 * sn);
    ko[qk + d + 32] = (bf16_t)(k1 * sn + k2 * cs);
    long vb = (long)(b * Hh + h) * HDq * Tq;
    vto[vb + (long)d * Tq + t]        = (bf16_t)base[2 * HDq + d];
    vto[vb + (long)(d + 32) * Tq + t] = (bf16_t)base[2 * HDq + d + 32];
}

// ---------------- flash attention ------------------
// one wave per 16-row Q tile; ctx written bf16 [B,T,H*HD]
__global__ __launch_bounds__(32) void flash_attn(
    const bf16_t* __restrict__ Q, const bf16_t* __restrict__ Km,
    const bf16_t* __restrict__ Vt, bf16_t* __restrict__ ctx)
{
    __shared__ __attribute__((aligned(16))) bf16_t pLds[16 * 32];
    int qt = blockIdx.x, h = blockIdx.y, b = blockIdx.z;
    int q0 = qt * 16;
    int lane = threadIdx.x;
    int lo = lane & 15, hi = lane >> 4;
    long bh = (long)(b * Hh + h);
    const bf16_t* qp = Q + (bh * Tq + q0) * HDq;
    const bf16_t* kp = Km + bh * Tq * HDq;
    const bf16_t* vp = Vt + bh * (long)HDq * Tq;

    // Q A-fragments: d 0..31 and 32..63
    v16bf qa0, qa1;
    {
        const bf16_t* r = qp + (long)lo * HDq + 8 * hi;
        v8bf x0 = *(const v8bf*)(r),      x1 = *(const v8bf*)(r + 16);
        v8bf y0 = *(const v8bf*)(r + 32), y1 = *(const v8bf*)(r + 48);
#pragma unroll
        for (int i = 0; i < 8; i++) {
            qa0[i] = x0[i]; qa0[8 + i] = x1[i];
            qa1[i] = y0[i]; qa1[8 + i] = y1[i];
        }
    }

    float rm[8], rl[8];
#pragma unroll
    for (int j = 0; j < 8; j++) { rm[j] = -1e30f; rl[j] = 0.f; }
    v8f acc[4] = {};
    const float scale = 0.125f; // 1/sqrt(64)

    for (int kb = 0; kb < q0 + 16; kb += 32) {
        v8f S[2];
#pragma unroll
        for (int sub = 0; sub < 2; sub++) {
            // B fragment: rows are keys, K-dim is head dim (two halves)
            const bf16_t* kr = kp + (long)(kb + 16 * sub + lo) * HDq + 8 * hi;
            v16bf b0, b1;
            v8bf x0 = *(const v8bf*)(kr),      x1 = *(const v8bf*)(kr + 16);
            v8bf y0 = *(const v8bf*)(kr + 32), y1 = *(const v8bf*)(kr + 48);
#pragma unroll
            for (int i = 0; i < 8; i++) {
                b0[i] = x0[i]; b0[8 + i] = x1[i];
                b1[i] = y0[i]; b1[8 + i] = y1[i];
            }
            v8f z = {};
            z = __builtin_amdgcn_wmma_f32_16x16x32_bf16(false, qa0, false, b0, (short)0, z, false, false);
            z = __builtin_amdgcn_wmma_f32_16x16x32_bf16(false, qa1, false, b1, (short)0, z, false, false);
            S[sub] = z;
        }
        float pex[2][8];
#pragma unroll
        for (int j = 0; j < 8; j++) {
            int m = q0 + 8 * hi + j;
            float s0 = S[0][j] * scale;
            float s1 = S[1][j] * scale;
            if (kb + lo > m)      s0 = -1e30f;
            if (kb + 16 + lo > m) s1 = -1e30f;
            float mx = fmaxf(s0, s1);
            for (int o = 1; o < 16; o <<= 1) mx = fmaxf(mx, __shfl_xor(mx, o, 32));
            float mnew = fmaxf(rm[j], mx);
            float corr = __expf(rm[j] - mnew);
            float p0 = __expf(s0 - mnew), p1 = __expf(s1 - mnew);
            float ps = p0 + p1;
            for (int o = 1; o < 16; o <<= 1) ps += __shfl_xor(ps, o, 32);
            rl[j] = rl[j] * corr + ps;
            rm[j] = mnew;
#pragma unroll
            for (int s = 0; s < 4; s++) acc[s][j] = acc[s][j] * corr;
            pex[0][j] = p0; pex[1][j] = p1;
        }
        // C-layout -> A-layout transpose of P through LDS
#pragma unroll
        for (int sub = 0; sub < 2; sub++)
#pragma unroll
            for (int j = 0; j < 8; j++)
                pLds[(8 * hi + j) * 32 + 16 * sub + lo] = (bf16_t)pex[sub][j];
        __syncthreads();
        v16bf pf;
        {
            const bf16_t* r = pLds + lo * 32 + 8 * hi;
            v8bf x0 = *(const v8bf*)(r), x1 = *(const v8bf*)(r + 16);
#pragma unroll
            for (int i = 0; i < 8; i++) { pf[i] = x0[i]; pf[8 + i] = x1[i]; }
        }
        __syncthreads();
        // ctx += P @ V ; V transposed so B rows = head-dim, contiguous over keys
#pragma unroll
        for (int s = 0; s < 4; s++) {
            const bf16_t* vr = vp + (long)(s * 16 + lo) * Tq + kb + 8 * hi;
            v16bf vf;
            v8bf x0 = *(const v8bf*)(vr), x1 = *(const v8bf*)(vr + 16);
#pragma unroll
            for (int i = 0; i < 8; i++) { vf[i] = x0[i]; vf[8 + i] = x1[i]; }
            acc[s] = __builtin_amdgcn_wmma_f32_16x16x32_bf16(
                false, pf, false, vf, (short)0, acc[s], false, false);
        }
    }
    // normalize and store ctx bf16 [B,T,H,HD]
#pragma unroll
    for (int s = 0; s < 4; s++) {
        int d = s * 16 + lo;
#pragma unroll
        for (int j = 0; j < 8; j++) {
            int m = q0 + 8 * hi + j;
            float v = acc[s][j] / rl[j];
            ctx[(long)(b * Tq + m) * Cq + h * HDq + d] = (bf16_t)v;
        }
    }
}

// ---------------- elementwise ----------------------
__global__ __launch_bounds__(256) void add_residual(const float* __restrict__ a,
                                                    const float* __restrict__ b,
                                                    float* __restrict__ o, long n) {
    long i = (long)blockIdx.x * 256 + threadIdx.x;
    if (i < n) o[i] = a[i] + b[i];
}

// ---------------- gating ---------------------------
// one wave per token: logits, softmax over 8, top-2 -> scores (gate or -1)
__global__ __launch_bounds__(32) void gate_topk(const bf16_t* __restrict__ ln2,
                                                const float* __restrict__ gw,
                                                float* __restrict__ scores) {
    int i = blockIdx.x;
    int lane = threadIdx.x;
    float accv[Eq] = {};
    const bf16_t* xr = ln2 + (long)i * Cq;
    for (int c = lane; c < Cq; c += 32) {
        float xv = (float)xr[c];
#pragma unroll
        for (int e = 0; e < Eq; e++) accv[e] += xv * gw[e * Cq + c];
    }
#pragma unroll
    for (int e = 0; e < Eq; e++)
        for (int o = 16; o >= 1; o >>= 1) accv[e] += __shfl_xor(accv[e], o, 32);
    if (lane == 0) {
        float mx = accv[0];
        for (int e = 1; e < Eq; e++) mx = fmaxf(mx, accv[e]);
        float g[Eq]; float se = 0.f;
        for (int e = 0; e < Eq; e++) { g[e] = __expf(accv[e] - mx); se += g[e]; }
        float inv = 1.0f / se;
        for (int e = 0; e < Eq; e++) g[e] *= inv;
        int i1 = 0;
        for (int e = 1; e < Eq; e++) if (g[e] > g[i1]) i1 = e;
        int i2 = -1;
        for (int e = 0; e < Eq; e++) { if (e == i1) continue; if (i2 < 0 || g[e] > g[i2]) i2 = e; }
        for (int e = 0; e < Eq; e++)
            scores[(long)i * Eq + e] = (e == i1 || e == i2) ? g[e] : -1.0f;
    }
}

__global__ __launch_bounds__(256) void init_caps(int* __restrict__ ci, float* __restrict__ cw) {
    int i = blockIdx.x * 256 + threadIdx.x;
    if (i < Eq * CAPq) { ci[i] = -1; cw[i] = 0.f; }
}

// rank-by-count capacity assignment (matches score-ordered top_k, stable ties)
__global__ __launch_bounds__(256) void capacity_rank(const float* __restrict__ scores,
                                                     int* __restrict__ cap_idx,
                                                     float* __restrict__ cap_w) {
    int e = blockIdx.y;
    int i = blockIdx.x;
    float si = scores[(long)i * Eq + e];
    __shared__ int red[8];
    int cnt = 0;
    for (int j = threadIdx.x; j < NTOK; j += 256) {
        float sj = scores[(long)j * Eq + e];
        cnt += (sj > si) || (sj == si && j < i);
    }
    for (int o = 16; o >= 1; o >>= 1) cnt += __shfl_xor(cnt, o, 32);
    int wave = threadIdx.x >> 5;
    if ((threadIdx.x & 31) == 0) red[wave] = cnt;
    __syncthreads();
    if (threadIdx.x == 0) {
        int tot = 0;
        for (int w = 0; w < 8; w++) tot += red[w];
        if (si > 0.f && tot < CAPq) {
            cap_idx[e * CAPq + tot] = i;
            cap_w[e * CAPq + tot] = si;
        }
    }
}

__global__ __launch_bounds__(256) void gather_toks(const int* __restrict__ cap_idx,
                                                   const bf16_t* __restrict__ ln2,
                                                   bf16_t* __restrict__ toks) {
    long slot = blockIdx.x;
    int idx = cap_idx[slot];
    for (int c = threadIdx.x; c < Cq; c += 256)
        toks[slot * Cq + c] = (idx >= 0) ? ln2[(long)idx * Cq + c] : (bf16_t)0.0f;
}

__global__ __launch_bounds__(256) void scatter_moe(const int* __restrict__ cap_idx,
                                                   const float* __restrict__ cap_w,
                                                   const float* __restrict__ eout,
                                                   float* __restrict__ out) {
    long slot = blockIdx.x;
    int idx = cap_idx[slot];
    if (idx < 0) return;
    float w = cap_w[slot];
    for (int c = threadIdx.x; c < Cq; c += 256)
        atomicAdd(&out[(long)idx * Cq + c], eout[slot * Cq + c] * w);
}

// ---------------- launcher -------------------------
extern "C" void kernel_launch(void* const* d_in, const int* in_sizes, int n_in,
                              void* d_out, int out_size, void* d_ws, size_t ws_size,
                              hipStream_t stream) {
    const float* x      = (const float*)d_in[0];
    const float* ln1_w  = (const float*)d_in[1];
    const float* ln1_b  = (const float*)d_in[2];
    const float* qkv_w  = (const float*)d_in[3];
    const float* proj_w = (const float*)d_in[4];
    const float* ln2_w  = (const float*)d_in[5];
    const float* ln2_b  = (const float*)d_in[6];
    const float* gate_w = (const float*)d_in[7];
    const float* w1     = (const float*)d_in[8];
    const float* b1     = (const float*)d_in[9];
    const float* w2     = (const float*)d_in[10];
    const float* b2     = (const float*)d_in[11];
    float* out = (float*)d_out;

    char* ws = (char*)d_ws;
    size_t off = 0;
    auto alloc = [&](size_t bytes) -> char* {
        char* p = ws + off;
        off = (off + bytes + 255) & ~(size_t)255;
        return p;
    };

    bf16_t* qkvw_bf = (bf16_t*)alloc((size_t)3 * Cq * Cq * 2);
    bf16_t* projw_bf = (bf16_t*)alloc((size_t)Cq * Cq * 2);
    bf16_t* w1t_bf  = (bf16_t*)alloc((size_t)Eq * Cq * DFFq * 2);
    bf16_t* w2t_bf  = (bf16_t*)alloc((size_t)Eq * DFFq * Cq * 2);
    bf16_t* ln1_bf  = (bf16_t*)alloc((size_t)NTOK * Cq * 2);
    float*  qkv_f   = (float*) alloc((size_t)NTOK * 3 * Cq * 4);
    bf16_t* q_bf    = (bf16_t*)alloc((size_t)NTOK * Cq * 2);
    bf16_t* k_bf    = (bf16_t*)alloc((size_t)NTOK * Cq * 2);
    bf16_t* vT_bf   = (bf16_t*)alloc((size_t)NTOK * Cq * 2);
    bf16_t* ctx_bf  = (bf16_t*)alloc((size_t)NTOK * Cq * 2);
    float*  proj_f  = (float*) alloc((size_t)NTOK * Cq * 4);
    bf16_t* ln2_bf  = (bf16_t*)alloc((size_t)NTOK * Cq * 2);
    float*  scores  = (float*) alloc((size_t)NTOK * Eq * 4);
    int*    cap_idx = (int*)   alloc((size_t)Eq * CAPq * 4);
    float*  cap_w   = (float*) alloc((size_t)Eq * CAPq * 4);
    bf16_t* toks_bf = (bf16_t*)alloc((size_t)Eq * CAPq * Cq * 2);
    bf16_t* hid_bf  = (bf16_t*)alloc((size_t)Eq * CAPq * DFFq * 2);
    float*  eout_f  = (float*) alloc((size_t)Eq * CAPq * Cq * 4);

    // 1) weight conversion (bf16, with transpose for w1/w2 so B is [n][k])
    {
        long n = (long)3 * Cq * Cq;
        cvt_bf16<<<dim3((n + 255) / 256), dim3(256), 0, stream>>>(qkv_w, qkvw_bf, n);
        n = (long)Cq * Cq;
        cvt_bf16<<<dim3((n + 255) / 256), dim3(256), 0, stream>>>(proj_w, projw_bf, n);
        cvt_bf16_tr<<<dim3(DFFq / 32, Cq / 32, Eq), dim3(256), 0, stream>>>(w1, w1t_bf, Cq, DFFq);
        cvt_bf16_tr<<<dim3(Cq / 32, DFFq / 32, Eq), dim3(256), 0, stream>>>(w2, w2t_bf, DFFq, Cq);
    }

    // 2) LN1
    layernorm_bf16<<<dim3(NTOK), dim3(256), 0, stream>>>(x, ln1_w, ln1_b, ln1_bf);

    // 3) QKV GEMM: [4096,1024] x [3072,1024]^T -> f32 [4096,3072]
    gemm_bf16<0><<<dim3(3 * Cq / 128, NTOK / 128, 1), dim3(256), 0, stream>>>(
        ln1_bf, qkvw_bf, nullptr, qkv_f, NTOK, 3 * Cq, Cq, 0, 0, 0, 0);

    // 4) RoPE + split + V transpose
    rope_split<<<dim3(Tq, Hh, Bq), dim3(32), 0, stream>>>(qkv_f, q_bf, k_bf, vT_bf);

    // 5) causal flash attention, ctx bf16 [B,T,C]
    flash_attn<<<dim3(Tq / 16, Hh, Bq), dim3(32), 0, stream>>>(q_bf, k_bf, vT_bf, ctx_bf);

    // 6) proj GEMM -> f32
    gemm_bf16<0><<<dim3(Cq / 128, NTOK / 128, 1), dim3(256), 0, stream>>>(
        ctx_bf, projw_bf, nullptr, proj_f, NTOK, Cq, Cq, 0, 0, 0, 0);

    // 7) h = x + attn ; stored in d_out (overwritten every call -> graph-replay safe)
    add_residual<<<dim3((NTOK * Cq) / 256), dim3(256), 0, stream>>>(
        x, proj_f, out, (long)NTOK * Cq);

    // 8) LN2
    layernorm_bf16<<<dim3(NTOK), dim3(256), 0, stream>>>(out, ln2_w, ln2_b, ln2_bf);

    // 9) gating + top-2 scores
    gate_topk<<<dim3(NTOK), dim3(32), 0, stream>>>(ln2_bf, gate_w, scores);

    // 10) capacity assignment (score-ordered, stable)
    init_caps<<<dim3((Eq * CAPq + 255) / 256), dim3(256), 0, stream>>>(cap_idx, cap_w);
    capacity_rank<<<dim3(NTOK, Eq), dim3(256), 0, stream>>>(scores, cap_idx, cap_w);

    // 11) gather tokens per expert slot
    gather_toks<<<dim3(Eq * CAPq), dim3(256), 0, stream>>>(cap_idx, ln2_bf, toks_bf);

    // 12) expert FFN GEMMs (batched over experts)
    gemm_bf16<2><<<dim3(DFFq / 128, CAPq / 128, Eq), dim3(256), 0, stream>>>(
        toks_bf, w1t_bf, b1, hid_bf, CAPq, DFFq, Cq,
        (long)CAPq * Cq, (long)DFFq * Cq, (long)DFFq, (long)CAPq * DFFq);
    gemm_bf16<1><<<dim3(Cq / 128, CAPq / 128, Eq), dim3(256), 0, stream>>>(
        hid_bf, w2t_bf, b2, eout_f, CAPq, Cq, DFFq,
        (long)CAPq * DFFq, (long)Cq * DFFq, (long)Cq, (long)CAPq * Cq);

    // 13) weighted scatter-add into d_out
    scatter_moe<<<dim3(Eq * CAPq), dim3(256), 0, stream>>>(cap_idx, cap_w, eout_f, out);
}
